// LorentzRankingLoss_15049565405360
// MI455X (gfx1250) — matrix operand order repaired and checked by reading, not compiled, not run on hardware
//
#include <hip/hip_runtime.h>
#include <hip/hip_bf16.h>

// ---------------- problem constants (reference shapes are fixed) -------------
#define NB     2            // batch
#define NCH    32           // channels C
#define NC     105          // num classes
#define NCP    112          // padded to 7*16
#define NTIL   7            // NCP/16
#define SPC    64           // samples per class S
#define KANCH  (NC*SPC)     // 6720 anchors
#define NTILES (KANCH/16)   // 420 anchor tiles
#define WPB    8            // waves per block in fused kernel
#define NNEG   8
#define MARGIN 0.1f
// T = 2.0 * (0.1/2.0)^((10-5)/(50-5)) = 2*0.05^(1/9)
#define INV_T  (1.0f/1.4337542f)

typedef __attribute__((ext_vector_type(16))) _Float16 v16h;
typedef __attribute__((ext_vector_type(8)))  float    v8f;
typedef __attribute__((ext_vector_type(4)))  unsigned v4u;
typedef __attribute__((ext_vector_type(8)))  int      v8i;
typedef __attribute__((ext_vector_type(4)))  int      v4i;

__device__ __forceinline__ unsigned pcg_hash(unsigned x) {
  x ^= x >> 17; x *= 0xed5ad4bbu;
  x ^= x >> 11; x *= 0xac4c1b51u;
  x ^= x >> 15; x *= 0x31848babu;
  x ^= x >> 14; return x;
}

__device__ __forceinline__ float gumbel_noise(int row, int col) {
  unsigned h = pcg_hash((unsigned)row * 1664525u + (unsigned)col * 1013904223u + 0x9E3779B9u);
  float u = ((float)(h >> 8) + 0.5f) * (1.0f / 16777216.0f); // (0,1)
  return -__logf(-__logf(u));
}

// ------------------------- 1: zero counters ---------------------------------
__global__ void zero_kernel(int* counts) {
  int i = threadIdx.x;
  if (i < NC) counts[i] = 0;
}

// ------------------------- 2: label histogram -------------------------------
__global__ void hist_kernel(const int* __restrict__ labels, int n, int* __restrict__ counts) {
  __shared__ int h[NC];
  for (int i = threadIdx.x; i < NC; i += blockDim.x) h[i] = 0;
  __syncthreads();
  int stride = gridDim.x * blockDim.x;
  for (int i = blockIdx.x * blockDim.x + threadIdx.x; i < n; i += stride) {
    __builtin_prefetch(&labels[i + stride], 0, 1); // global_prefetch_b8
    atomicAdd(&h[labels[i]], 1);
  }
  __syncthreads();
  for (int i = threadIdx.x; i < NC; i += blockDim.x)
    if (h[i]) atomicAdd(&counts[i], h[i]);
}

// ---- 3: label table -> f16, pre-swizzled into the WMMA B-fragment layout ----
// B tile t, lane l (col = l&15, half = l>>4), elem i  <->  class (t*16+col), half = (k>>3)&1, i = (k&7) + 8*(k>>4)
__global__ void labelprep_kernel(const float* __restrict__ label_emb,
                                 _Float16* __restrict__ Lswz, float* __restrict__ tL) {
  int c = threadIdx.x;
  if (c >= NCP) return;
  int tile = c >> 4, col = c & 15;
  float ss = 0.f;
  for (int k = 0; k < NCH; ++k) {
    float v = (c < NC) ? label_emb[c * NCH + k] : 0.f;
    int lanep = col + 16 * ((k >> 3) & 1);
    int i     = (k & 7) + 8 * (k >> 4);
    Lswz[(tile * 32 + lanep) * 16 + i] = (_Float16)v;
    ss += v * v;
  }
  tL[c] = sqrtf(1.0f + ss); // curv = 1
}

// ------------------ 4: deterministic per-slot class sampling ----------------
__global__ void sample_kernel(const int* __restrict__ labels, const int* __restrict__ counts,
                              int n, int* __restrict__ sampled) {
  int s = blockIdx.x * blockDim.x + threadIdx.x;
  if (s >= KANCH) return;
  int cls = s >> 6; // /SPC
  if (counts[cls] == 0) { sampled[s] = 0; return; }
  unsigned pos = pcg_hash((unsigned)s * 2654435761u + 12345u) % (unsigned)n;
  for (int j = 0; j < n; ++j) {
    if (labels[pos] == cls) { sampled[s] = (int)pos; return; }
    pos += 97u;
    if (pos >= (unsigned)n) pos -= (unsigned)n;
  }
  sampled[s] = 0;
}

// ---- 5: gather anchors -> f16 pre-swizzled A-fragments + Lorentz time ------
// A tile: lane l (row = l&15, half = l>>4), elem i <-> channel k:
//   half = (k>>3)&1,  i = (k&7) + 8*(k>>4)
__global__ void gather_kernel(const float* __restrict__ vox, const int* __restrict__ sampled,
                              _Float16* __restrict__ Aswz, float* __restrict__ tA, int dhw) {
  int wave = (int)((blockIdx.x * blockDim.x + threadIdx.x) >> 5);
  int lane = threadIdx.x & 31; // = channel
  if (wave >= KANCH) return;
  int idx = sampled[wave];
  int b  = idx / dhw;
  int sp = idx - b * dhw;
  float x = vox[(size_t)(b * NCH + lane) * (size_t)dhw + (size_t)sp];
  float ss = x * x;
  #pragma unroll
  for (int m = 16; m >= 1; m >>= 1) ss += __shfl_xor(ss, m, 32);
  int tile = wave >> 4, row = wave & 15;
  int lanep = row + 16 * ((lane >> 3) & 1);
  int i     = (lane & 7) + 8 * (lane >> 4);
  Aswz[((size_t)tile * 32 + lanep) * 16 + i] = (_Float16)x;
  if (lane == 0) tA[wave] = sqrtf(1.0f + ss);
}

// ------- 6: TDM B->LDS + fused WMMA + acosh + Gumbel top-8 + triplet --------
__global__ __launch_bounds__(32 * WPB)
void fused_kernel(const _Float16* __restrict__ Aswz, const _Float16* __restrict__ Lswz,
                  const float* __restrict__ tA, const float* __restrict__ tL,
                  const int* __restrict__ counts, float* __restrict__ partials) {
  __shared__ __align__(32) _Float16 ldsB[NCP * NCH]; // 3584 halfs = 7 KB
  const int wv = threadIdx.x >> 5;
  const int l  = threadIdx.x & 31;

  // Wave 0: one TDM transfer of the whole swizzled B table into LDS.
  if (wv == 0) {
    unsigned lds_base = (unsigned)(size_t)(&ldsB[0]);
    unsigned long long ga = (unsigned long long)(size_t)Lswz;
    // D# group0: count=1 | lds_addr | 57b global_addr | type=2
    v4u g0 = { 1u, lds_base, (unsigned)ga,
               (unsigned)((ga >> 32) & 0x01FFFFFFu) | (2u << 30) };
    // D# group1: data_size=1(2B); tensor_dim0=tile_dim0=3584; dim1=tile_dim1=1;
    //            tensor_dim0_stride=3584
    v8i g1 = { (int)(1u << 16),              // [17:16] data_size=1, wg_mask=0
               (int)((unsigned)(NCP * NCH) << 16), // [79:48] tensor_dim0 (low 16)
               (int)(1u << 16),              // [111:80] tensor_dim1 = 1
               (int)((unsigned)(NCP * NCH) << 16), // [127:112] tile_dim0
               1,                            // [143:128] tile_dim1 = 1
               (int)(NCP * NCH),             // [207:160] tensor_dim0_stride
               0, 0 };
    v4i gz4 = { 0, 0, 0, 0 };
    v8i gz8 = { 0, 0, 0, 0, 0, 0, 0, 0 };
    __builtin_amdgcn_tensor_load_to_lds(g0, g1, gz4, gz4, gz8, 0);
    __builtin_amdgcn_s_wait_tensorcnt((short)0);
  }
  __syncthreads();

  const int tile = blockIdx.x * WPB + wv;
  if (tile >= NTILES) return;
  const int half = l >> 4;
  const int lc   = l & 15;
  const int row0 = tile * 16;

  // A fragment: one contiguous 32B load per lane (pre-swizzled)
  const v16h a = *(const v16h*)(Aswz + ((size_t)tile * 32 + l) * 16);

  float4 t0 = *(const float4*)&tA[row0 + half * 8];
  float4 t1 = *(const float4*)&tA[row0 + half * 8 + 4];
  float ta_r[8] = { t0.x, t0.y, t0.z, t0.w, t1.x, t1.y, t1.z, t1.w };

  // B fragments from LDS, then 7 back-to-back WMMAs (hides WMMA->VALU hazards)
  v16h bfrag[NTIL];
  #pragma unroll
  for (int t = 0; t < NTIL; ++t)
    bfrag[t] = *(const v16h*)(&ldsB[(t * 32 + l) * 16]);

  v8f acc[NTIL];
  #pragma unroll
  for (int t = 0; t < NTIL; ++t) {
    v8f c = {};
    acc[t] = __builtin_amdgcn_wmma_f32_16x16x32_f16(false, a, false, bfrag[t],
                                                    (short)0, c, false, false);
  }

  float dist[NTIL][8];
  #pragma unroll
  for (int t = 0; t < NTIL; ++t) {
    float tl = tL[t * 16 + lc];
    #pragma unroll
    for (int r = 0; r < 8; ++r) {
      float inner = fmaxf(ta_r[r] * tl - acc[t][r], 1.0f + 1e-7f);
      dist[t][r] = __logf(inner + __fsqrt_rn(inner * inner - 1.0f)); // acosh
    }
  }

  // ---- per-row Gumbel top-8 mining (cross-lane within each 16-lane half) ----
  float rowAcc = 0.f; // lane lc==0 of each half accumulates its 8 rows
  #pragma unroll
  for (int r = 0; r < 8; ++r) {
    const int gk   = row0 + half * 8 + r;
    const int cls  = gk >> 6;  // class = k / S
    const int slot = gk & 63;
    const float w  = (slot < counts[cls]) ? 1.0f : 0.0f;

    // d_pos: broadcast from owner lane of the positive class column
    const int pt = cls >> 4, pl = cls & 15;
    float ptmp = dist[0][r];
    #pragma unroll
    for (int t = 1; t < NTIL; ++t) if (pt == t) ptmp = dist[t][r];
    const float dpos = __shfl(ptmp, (half << 4) + pl, 32);

    float dr[NTIL], lw[NTIL];
    #pragma unroll
    for (int t = 0; t < NTIL; ++t) {
      dr[t] = dist[t][r];
      int col = t * 16 + lc;
      bool isneg = (col < NC) && (col != cls);
      lw[t] = isneg ? (-dr[t] * INV_T + gumbel_noise(gk, col)) : -1e30f;
    }

    float trip = 0.f;
    #pragma unroll 1
    for (int it = 0; it < NNEG; ++it) {
      float bv = -1e30f; int bt = 0;
      #pragma unroll
      for (int t = 0; t < NTIL; ++t) if (lw[t] > bv) { bv = lw[t]; bt = t; }
      float v = bv; int id = lc * 8 + bt;
      #pragma unroll
      for (int m = 8; m >= 1; m >>= 1) {
        float ov = __shfl_xor(v, m, 16);
        int  oid = __shfl_xor(id, m, 16);
        if (ov > v || (ov == v && oid < id)) { v = ov; id = oid; }
      }
      const int wl = id >> 3, wt = id & 7;     // uniform across the half
      float dtmp = dr[0];
      #pragma unroll
      for (int t = 1; t < NTIL; ++t) if (wt == t) dtmp = dr[t];
      const float dneg = __shfl(dtmp, (half << 4) + wl, 32);
      #pragma unroll
      for (int t = 0; t < NTIL; ++t) if (t == wt && lc == wl) lw[t] = -1e30f;
      trip += fmaxf(MARGIN + dpos - dneg, 0.f);
    }
    if (lc == 0) rowAcc += trip * w;
  }
  const float other = __shfl(rowAcc, 16, 32);
  if (l == 0) partials[tile] = rowAcc + other;
}

// ------------------------- 7: deterministic finalize ------------------------
__global__ void finalize_kernel(const float* __restrict__ partials, int nb,
                                const int* __restrict__ counts, float* __restrict__ out) {
  if (blockIdx.x == 0 && threadIdx.x == 0) {
    float s = 0.f;
    for (int i = 0; i < nb; ++i) s += partials[i];  // fixed order
    int v = 0;
    for (int c = 0; c < NC; ++c) v += (counts[c] < SPC) ? counts[c] : SPC;
    float denom = (float)v * (float)NNEG;
    out[0] = s / fmaxf(denom, 1.0f);
  }
}

// ----------------------------------------------------------------------------
extern "C" void kernel_launch(void* const* d_in, const int* in_sizes, int n_in,
                              void* d_out, int out_size, void* d_ws, size_t ws_size,
                              hipStream_t stream) {
  const float* vox       = (const float*)d_in[0]; // (B, C, D, H, W)
  const int*   labels    = (const int*)d_in[1];   // (B, D, H, W)
  const float* label_emb = (const float*)d_in[2]; // (NC, C)
  float* out = (float*)d_out;

  const int Nvox = in_sizes[1];                   // 2,097,152
  const int dhw  = Nvox / NB;

  // workspace layout (256B-aligned)
  char* ws = (char*)d_ws;
  size_t off = 0;
  auto take = [&](size_t bytes) { size_t r = off; off = (off + bytes + 255) & ~(size_t)255; return r; };
  int*      counts  = (int*)     (ws + take(NC * sizeof(int)));
  int*      sampled = (int*)     (ws + take(KANCH * sizeof(int)));
  float*    tA      = (float*)   (ws + take(KANCH * sizeof(float)));
  float*    tL      = (float*)   (ws + take(NCP * sizeof(float)));
  _Float16* Aswz    = (_Float16*)(ws + take((size_t)KANCH * NCH * sizeof(_Float16)));
  _Float16* Lswz    = (_Float16*)(ws + take((size_t)NCP * NCH * sizeof(_Float16)));
  float*    partials = (float*)  (ws + take(NTILES * sizeof(float)));
  (void)ws_size; (void)n_in; (void)out_size;

  zero_kernel<<<1, 128, 0, stream>>>(counts);
  hist_kernel<<<1024, 256, 0, stream>>>(labels, Nvox, counts);
  labelprep_kernel<<<1, 128, 0, stream>>>(label_emb, Lswz, tL);
  sample_kernel<<<(KANCH + 255) / 256, 256, 0, stream>>>(labels, counts, Nvox, sampled);
  gather_kernel<<<(KANCH * 32 + 255) / 256, 256, 0, stream>>>(vox, sampled, Aswz, tA, dhw);
  fused_kernel<<<(NTILES + WPB - 1) / WPB, 32 * WPB, 0, stream>>>(Aswz, Lswz, tA, tL,
                                                                  counts, partials);
  finalize_kernel<<<1, 32, 0, stream>>>(partials, NTILES, counts, out);
}
